// Conv2DAttentionBlock_23287312679379
// MI455X (gfx1250) — matrix operand order repaired and checked by reading, not compiled, not run on hardware
//
// Conv2D attention block for MI455X (gfx1250), bf16 WMMA + flash-attention tiling.
// Pipeline: [1] QKV projection (f32->bf16; q,k stored transposed [b][h][n][d],
//               v stored natural [b][h][d][m]) ->
//           [2] 4-head flash attention, Q/K/V tiles staged via TENSOR_LOAD_TO_LDS ->
//           [3] output projection + bias + residual (fp32 out).
// Workspace requirement: 4 * 16*512*1024 * 2 bytes = 64 MiB.
#include <hip/hip_runtime.h>
#include <stdint.h>

typedef __attribute__((ext_vector_type(16))) __bf16 v16bf;
typedef __attribute__((ext_vector_type(8)))  float  v8f;
typedef __attribute__((ext_vector_type(4)))  unsigned int v4u;
typedef __attribute__((ext_vector_type(8)))  int    v8i;
typedef __attribute__((ext_vector_type(4)))  int    v4i;

union V16 { v16bf v; unsigned short u[16]; };

static constexpr int BATCH  = 16;
static constexpr int DIM    = 128;
static constexpr int HEADS  = 4;
static constexpr int EXPAND = 512;
static constexpr int HW     = 1024;
static constexpr size_t QKV_ONE = (size_t)BATCH * EXPAND * HW;   // elems per q/k/v tensor

#if defined(__has_builtin)
#  if __has_builtin(__builtin_amdgcn_tensor_load_to_lds)
#    define HAS_TDM 1
#  endif
#endif
#ifndef HAS_TDM
#  define HAS_TDM 0
#endif

__device__ __forceinline__ unsigned short f2bf(float f) {
  union { float f; uint32_t u; } c; c.f = f;
  uint32_t r = c.u + 0x7FFFu + ((c.u >> 16) & 1u);   // round-to-nearest-even
  return (unsigned short)(r >> 16);
}

#if HAS_TDM
// Issue a TDM load of a 2D bf16 tile (rows x cols, cols contiguous in memory)
// into LDS at byte offset lds_off, padding each row per pad codes (ISA 8.4).
__device__ __forceinline__ void tdm_load_2d_bf16(
    const unsigned short* gsrc, uint32_t lds_off,
    uint32_t rows, uint32_t cols,
    uint32_t tensor_d0, uint32_t tensor_d1, uint32_t row_stride_elems,
    uint32_t pad_interval_code, uint32_t pad_amount_code) {
  const uint64_t ga = (uint64_t)(uintptr_t)gsrc;
  union { v4u v; unsigned int u[4]; } g0;
  g0.u[0] = 1u;                                    // count=1, user descriptor
  g0.u[1] = lds_off;                               // lds_addr (bytes)
  g0.u[2] = (unsigned int)ga;                      // global_addr[31:0]
  g0.u[3] = (unsigned int)((ga >> 32) & 0x01FFFFFFu) | (2u << 30); // addr[56:32], type=2
  union { v8i v; uint64_t q[4]; } g1;
  const uint64_t flags = (1ull << 16)              // data_size = 1 -> 2 bytes
                       | (1ull << 20)              // pad_enable
                       | ((uint64_t)pad_interval_code << 22)
                       | ((uint64_t)pad_amount_code << 25);
  g1.q[0] = flags | ((uint64_t)(tensor_d0 & 0xFFFFu) << 48);
  g1.q[1] = ((uint64_t)(tensor_d0 >> 16) & 0xFFFFu)
          | (((uint64_t)tensor_d1 & 0xFFFFu) << 16)
          | (((uint64_t)(tensor_d1 >> 16) & 0xFFFFu) << 32)
          | ((uint64_t)(cols & 0xFFFFu) << 48);    // tile_dim0
  g1.q[2] = (uint64_t)(rows & 0xFFFFu)             // tile_dim1 (tile_dim2 = 0)
          | ((uint64_t)row_stride_elems << 32);    // tensor_dim0_stride[31:0]
  g1.q[3] = 0;                                     // stride0 hi / dim1 stride
  v4i z4 = {};
#if __clang_major__ >= 23
  v8i z8 = {};
  __builtin_amdgcn_tensor_load_to_lds(g0.v, g1.v, z4, z4, z8, 0);
#else
  __builtin_amdgcn_tensor_load_to_lds(g0.v, g1.v, z4, z4, 0);
#endif
}
#endif

// A-matrix fragment (16xK=32 bf16) from row-major LDS tile [16][stride].
__device__ __forceinline__ v16bf lds_frag_a(const unsigned short* base, int stride) {
  const int lane = threadIdx.x & 31;
  const unsigned short* row = base + (lane & 15) * stride + (lane >> 4) * 8;
  V16 a;
#pragma unroll
  for (int e = 0; e < 8; ++e) a.u[e] = row[e];
#pragma unroll
  for (int e = 0; e < 8; ++e) a.u[8 + e] = row[16 + e];
  return a.v;
}

// B-matrix fragment (K=32 x N=16 bf16) from Bt LDS tile stored [N][K] row-major.
__device__ __forceinline__ v16bf lds_frag_b(const unsigned short* base, int stride) {
  const int lane = threadIdx.x & 31;
  const unsigned short* row = base + (lane & 15) * stride + (lane >> 4) * 8;
  V16 b;
#pragma unroll
  for (int e = 0; e < 8; ++e) b.u[e] = row[e];
#pragma unroll
  for (int e = 0; e < 8; ++e) b.u[8 + e] = row[16 + e];
  return b.v;
}

#define WMMA_BF16(A, B, C) \
  __builtin_amdgcn_wmma_f32_16x16x32_bf16(false, (A), false, (B), (short)0, (C), false, false)

// ---------------------------------------------------------------------------
// Kernel 1: q/k/v[o,n] = W[o,c] @ x[b,c,n] (+bv for v), bf16 results.
// q,k stored TRANSPOSED as [b][h][n][d] (d contiguous) so the attention kernel
// can fetch Q/K tiles with straight TDM copies; v stored natural [b][h][d][m].
// ---------------------------------------------------------------------------
__global__ void __launch_bounds__(256)
qkv_proj_kernel(const float* __restrict__ x,
                const float* __restrict__ Wq, const float* __restrict__ Wk,
                const float* __restrict__ Wv, const float* __restrict__ bv,
                unsigned short* __restrict__ qkv) {
  const int nt   = blockIdx.x;            // 16 n-tiles
  const int b    = blockIdx.y;            // batch
  const int wsel = blockIdx.z >> 3;       // 0=q 1=k 2=v
  const int mt   = blockIdx.z & 7;        // 8 m-tiles of 64
  const float* W = (wsel == 0) ? Wq : ((wsel == 1) ? Wk : Wv);

  __shared__ __align__(16) unsigned short ldsA[64][40];  // [o][c]
  __shared__ __align__(16) unsigned short ldsB[64][40];  // [n][c] Bt layout

  const int tid = threadIdx.x;
  const int w   = tid >> 5;
  const int wm  = w >> 1, wn = w & 1;
  const int o0  = mt * 64, n0 = nt * 64;

  v8f acc0 = {}, acc1 = {};
#pragma unroll
  for (int c0 = 0; c0 < DIM; c0 += 32) {
    if (c0 + 32 < DIM) __builtin_prefetch(&W[(size_t)o0 * DIM + c0 + 32], 0, 1);
    for (int i = tid; i < 64 * 32; i += 256) {         // A: coalesced on c
      int r = i >> 5, cc = i & 31;
      ldsA[r][cc] = f2bf(W[(size_t)(o0 + r) * DIM + c0 + cc]);
    }
    for (int i = tid; i < 64 * 32; i += 256) {         // Bt: coalesced on n
      int cc = i >> 6, n = i & 63;
      ldsB[n][cc] = f2bf(x[((size_t)b * DIM + c0 + cc) * HW + n0 + n]);
    }
    __syncthreads();
    v16bf af = lds_frag_a(&ldsA[wm * 16][0], 40);
    v16bf b0 = lds_frag_b(&ldsB[wn * 32][0], 40);
    v16bf b1 = lds_frag_b(&ldsB[wn * 32 + 16][0], 40);
    acc0 = WMMA_BF16(af, b0, acc0);
    acc1 = WMMA_BF16(af, b1, acc1);
    __syncthreads();
  }

  const int c  = threadIdx.x & 15;
  const int hl = (threadIdx.x & 31) >> 4;
  const size_t reg = (size_t)wsel * QKV_ONE;
#pragma unroll
  for (int r = 0; r < 8; ++r) {
    const int o  = o0 + wm * 16 + r + 8 * hl;
    const int na = n0 + wn * 32 + c;
    if (wsel == 2) {                                   // v: natural [b][o][m]
      const float add = bv[o];
      const size_t base = reg + ((size_t)b * EXPAND + o) * HW;
      qkv[base + na]      = f2bf(acc0[r] + add);
      qkv[base + na + 16] = f2bf(acc1[r] + add);
    } else {                                           // q/k: transposed [b][h][n][d]
      const int hh = o >> 7, d = o & 127;
      const size_t base = reg + (size_t)(b * HEADS + hh) * HW * DIM + d;
      qkv[base + (size_t)na * DIM]        = f2bf(acc0[r]);
      qkv[base + (size_t)(na + 16) * DIM] = f2bf(acc1[r]);
    }
  }
}

// ---------------------------------------------------------------------------
// Kernel 2: flash attention per (batch, head). Block = 128 thr (4 waves),
// each wave owns 16 query rows end-to-end. Q/K/V tiles staged by the Tensor
// Data Mover (row padding via D# pad fields breaks LDS bank conflicts).
// ---------------------------------------------------------------------------
__global__ void __launch_bounds__(128)
flash_attn_kernel(const unsigned short* __restrict__ qb,   // [b][h][n][d]
                  const unsigned short* __restrict__ kb,   // [b][h][m][d]
                  const unsigned short* __restrict__ vb,   // [b][h][d][m]
                  unsigned short* __restrict__ ob) {       // [b][h][n][d]
  const int nt = blockIdx.x;   // 16 query tiles of 64
  const int h  = blockIdx.y;   // head
  const int b  = blockIdx.z;   // batch

  __shared__ __align__(16) unsigned short ldsQ[64][136];   // [n][d]  (A layout)
  __shared__ __align__(16) unsigned short ldsK[32][136];   // [m][d]  (Bt of K)
  __shared__ __align__(16) unsigned short ldsV[128][40];   // [dd][m] (Bt of Vt)
  __shared__ __align__(16) unsigned short ldsP[4][16][40]; // per-wave P re-layout

  const int tid  = threadIdx.x;
  const int lane = tid & 31;
  const int w    = tid >> 5;
  const int c    = lane & 15;
  const int hl   = lane >> 4;
  const int n0   = nt * 64;
  const float inv_norm = 0.08838834764831845f;             // 1/sqrt(128)

  const unsigned short* qhead = qb + (size_t)(b * HEADS + h) * HW * DIM;
  const unsigned short* khead = kb + (size_t)(b * HEADS + h) * HW * DIM;
  const unsigned short* vhead = vb + ((size_t)b * EXPAND + h * DIM) * HW;

#if HAS_TDM
  const uint32_t ldsQ_off = (uint32_t)(uintptr_t)&ldsQ[0][0];
  const uint32_t ldsK_off = (uint32_t)(uintptr_t)&ldsK[0][0];
  const uint32_t ldsV_off = (uint32_t)(uintptr_t)&ldsV[0][0];
  if (w == 0)    // Q tile: 64 rows x 128 cols, row pad 256B->+16B (codes 5,3)
    tdm_load_2d_bf16(qhead + (size_t)n0 * DIM, ldsQ_off, 64, DIM, DIM, HW, DIM, 5, 3);
#else
  for (int i = tid; i < 64 * DIM; i += 128) {
    int n = i >> 7, d = i & 127;
    ldsQ[n][d] = qhead[(size_t)(n0 + n) * DIM + d];
  }
#endif

  v8f acc[8]; float mrun[8], lrun[8];
#pragma unroll
  for (int t = 0; t < 8; ++t) acc[t] = (v8f){};
#pragma unroll
  for (int r = 0; r < 8; ++r) { mrun[r] = -1.0e30f; lrun[r] = 0.0f; }

  for (int mt = 0; mt < HW / 32; ++mt) {
    const int m0 = mt * 32;
#if HAS_TDM
    if (w == 0) { // K: 32x128 (codes 5,3); V: 128x32, row pad 64B->+16B (codes 3,3)
      tdm_load_2d_bf16(khead + (size_t)m0 * DIM, ldsK_off, 32, DIM, DIM, HW, DIM, 5, 3);
      tdm_load_2d_bf16(vhead + m0, ldsV_off, DIM, 32, HW, DIM, HW, 3, 3);
      __builtin_amdgcn_s_wait_tensorcnt(0);
    }
#else
    for (int i = tid; i < 32 * DIM; i += 128) {
      int m = i >> 7, d = i & 127;
      ldsK[m][d] = khead[(size_t)(m0 + m) * DIM + d];
    }
    for (int i = tid; i < DIM * 32; i += 128) {
      int dd = i >> 5, m = i & 31;
      ldsV[dd][m] = vhead[(size_t)dd * HW + m0 + m];
    }
#endif
    __syncthreads();

    // S tile: wave's 16 rows x 32 key cols, contraction over d=128
    v8f s0 = {}, s1 = {};
#pragma unroll
    for (int dk = 0; dk < DIM; dk += 32) {
      v16bf af = lds_frag_a(&ldsQ[w * 16][dk], 136);
      v16bf b0 = lds_frag_b(&ldsK[0][dk],  136);
      v16bf b1 = lds_frag_b(&ldsK[16][dk], 136);
      s0 = WMMA_BF16(af, b0, s0);
      s1 = WMMA_BF16(af, b1, s1);
    }

    // online softmax (rows r+8*hl; stats reduced over the 16-lane half)
    float rmax[8], rsum[8], p0[8], p1[8], alpha[8];
#pragma unroll
    for (int r = 0; r < 8; ++r) {
      s0[r] *= inv_norm; s1[r] *= inv_norm;
      rmax[r] = fmaxf(s0[r], s1[r]);
    }
#pragma unroll
    for (int r = 0; r < 8; ++r)
#pragma unroll
      for (int msk = 1; msk <= 8; msk <<= 1)
        rmax[r] = fmaxf(rmax[r], __shfl_xor(rmax[r], msk, 32));
#pragma unroll
    for (int r = 0; r < 8; ++r) {
      float mnew = fmaxf(mrun[r], rmax[r]);
      p0[r] = __expf(s0[r] - mnew);
      p1[r] = __expf(s1[r] - mnew);
      alpha[r] = __expf(mrun[r] - mnew);
      mrun[r] = mnew;
      rsum[r] = p0[r] + p1[r];
    }
#pragma unroll
    for (int r = 0; r < 8; ++r)
#pragma unroll
      for (int msk = 1; msk <= 8; msk <<= 1)
        rsum[r] += __shfl_xor(rsum[r], msk, 32);
#pragma unroll
    for (int r = 0; r < 8; ++r) {
      lrun[r] = lrun[r] * alpha[r] + rsum[r];
      ldsP[w][r + 8 * hl][c]      = f2bf(p0[r]);      // D-layout -> A-layout via LDS
      ldsP[w][r + 8 * hl][c + 16] = f2bf(p1[r]);
    }
#pragma unroll
    for (int t = 0; t < 8; ++t)
#pragma unroll
      for (int r = 0; r < 8; ++r) acc[t][r] *= alpha[r];

    // O += P @ Vt : same-wave LDS store->load is in-order (DScnt, ISA 7.3)
    v16bf pa = lds_frag_a(&ldsP[w][0][0], 40);
#pragma unroll
    for (int t = 0; t < 8; ++t) {
      v16bf bvf = lds_frag_b(&ldsV[t * 16][0], 40);
      acc[t] = WMMA_BF16(pa, bvf, acc[t]);
    }
    __syncthreads();
  }

  // normalize and store wsv[b][h][n][d] (bf16)
  const size_t obase = (size_t)(b * HEADS + h) * HW * DIM;
#pragma unroll
  for (int r = 0; r < 8; ++r) {
    const float invl = 1.0f / lrun[r];
    const int   n    = n0 + w * 16 + r + 8 * hl;
#pragma unroll
    for (int t = 0; t < 8; ++t)
      ob[obase + (size_t)n * DIM + t * 16 + c] = f2bf(acc[t][r] * invl);
  }
}

// ---------------------------------------------------------------------------
// Kernel 3: out[b,o,n] = x[b,o,n] + bt[o] + Wt[o,c] @ wsv_perm[b,c,n]
// wsv_perm(b,c,n): h=c>>7, c'=c&127, nq=c'*8 + (n>>7), d=n&127 -> wsv[b][h][nq][d]
// ---------------------------------------------------------------------------
__global__ void __launch_bounds__(256)
out_proj_kernel(const float* __restrict__ x, const float* __restrict__ Wt,
                const float* __restrict__ bt,
                const unsigned short* __restrict__ wsv,
                float* __restrict__ out) {
  const int nt = blockIdx.x;   // 16 n-tiles
  const int b  = blockIdx.y;
  const int mt = blockIdx.z;   // 2 m-tiles of 64 (o in 0..127)

  __shared__ __align__(16) unsigned short ldsA[64][40];
  __shared__ __align__(16) unsigned short ldsB[64][40];

  const int tid = threadIdx.x;
  const int w   = tid >> 5;
  const int wm  = w >> 1, wn = w & 1;
  const int o0  = mt * 64, n0 = nt * 64;

  v8f acc0 = {}, acc1 = {};
  for (int c0 = 0; c0 < EXPAND; c0 += 32) {
    if (c0 + 32 < EXPAND) __builtin_prefetch(&Wt[(size_t)o0 * EXPAND + c0 + 32], 0, 1);
    for (int i = tid; i < 64 * 32; i += 256) {
      int r = i >> 5, cc = i & 31;
      ldsA[r][cc] = f2bf(Wt[(size_t)(o0 + r) * EXPAND + c0 + cc]);
    }
    for (int i = tid; i < 64 * 32; i += 256) {   // permuted gather, coalesced on d
      int cc = i >> 6, n = i & 63;
      int cg = c0 + cc, hh = cg >> 7, cp = cg & 127;
      int nn = n0 + n, nq = cp * 8 + (nn >> 7), d = nn & 127;
      ldsB[n][cc] = wsv[(((size_t)b * HEADS + hh) * HW + nq) * DIM + d];
    }
    __syncthreads();
    v16bf af = lds_frag_a(&ldsA[wm * 16][0], 40);
    v16bf b0 = lds_frag_b(&ldsB[wn * 32][0], 40);
    v16bf b1 = lds_frag_b(&ldsB[wn * 32 + 16][0], 40);
    acc0 = WMMA_BF16(af, b0, acc0);
    acc1 = WMMA_BF16(af, b1, acc1);
    __syncthreads();
  }

  const int c  = threadIdx.x & 15;
  const int hl = (threadIdx.x & 31) >> 4;
#pragma unroll
  for (int r = 0; r < 8; ++r) {
    const int o  = o0 + wm * 16 + r + 8 * hl;
    const int n  = n0 + wn * 32 + c;
    const size_t idx = ((size_t)b * DIM + o) * HW + n;
    const float bias = bt[o];
    out[idx]      = acc0[r] + x[idx]      + bias;
    out[idx + 16] = acc1[r] + x[idx + 16] + bias;
  }
}

// ---------------------------------------------------------------------------
extern "C" void kernel_launch(void* const* d_in, const int* in_sizes, int n_in,
                              void* d_out, int out_size, void* d_ws, size_t ws_size,
                              hipStream_t stream) {
  (void)in_sizes; (void)n_in; (void)out_size; (void)ws_size;
  const float* x  = (const float*)d_in[0];
  const float* Wq = (const float*)d_in[1];
  const float* Wk = (const float*)d_in[2];
  const float* Wv = (const float*)d_in[3];
  const float* bv = (const float*)d_in[4];
  const float* Wt = (const float*)d_in[5];
  const float* bt = (const float*)d_in[6];

  unsigned short* qkv = (unsigned short*)d_ws;          // q_t | k_t | v regions
  unsigned short* wsv = qkv + 3 * QKV_ONE;              // [B][4][1024][128] bf16

  qkv_proj_kernel<<<dim3(HW / 64, BATCH, 3 * (EXPAND / 64)), 256, 0, stream>>>(
      x, Wq, Wk, Wv, bv, qkv);
  flash_attn_kernel<<<dim3(HW / 64, HEADS, BATCH), 128, 0, stream>>>(
      qkv, qkv + QKV_ONE, qkv + 2 * QKV_ONE, wsv);
  out_proj_kernel<<<dim3(HW / 64, BATCH, DIM / 64), 256, 0, stream>>>(
      x, Wt, bt, wsv, (float*)d_out);
}
